// Denoiser_60138132078816
// MI455X (gfx1250) — compile-verified, matrix operand order checked by reading, not run on hardware
//
#include <hip/hip_runtime.h>

#define NL 2048

typedef __attribute__((ext_vector_type(16))) __bf16 v16bf;
typedef __attribute__((ext_vector_type(8)))  float  v8f;

union FragU { uint4 q[2]; v16bf v; };
union PackU { uint4 q; unsigned short s[8]; };

__device__ __forceinline__ unsigned short f2bf(float f) {
  unsigned int u = __float_as_uint(f);
  unsigned int r = u + 0x7fffu + ((u >> 16) & 1u);   // round to nearest even
  return (unsigned short)(r >> 16);
}
__device__ __forceinline__ float sigmf(float x) { return 1.0f / (1.0f + __expf(-x)); }

// Async global->LDS copy of 16 bytes (gfx1250 ASYNCcnt path).
__device__ __forceinline__ void async_copy_b128(const void* gaddr, void* lds_ptr) {
  unsigned loff = (unsigned)(unsigned long long)lds_ptr;   // low 32b of generic = LDS offset
  asm volatile("global_load_async_to_lds_b128 %0, %1, off"
               :: "v"(loff), "v"(gaddr) : "memory");
}
__device__ __forceinline__ void wait_async0() {
  asm volatile("s_wait_asynccnt 0x0" ::: "memory");
}

// ---------------------------------------------------------------------------
// Row-wise weight normalization (_wnorm): one block per row
// ---------------------------------------------------------------------------
__global__ __launch_bounds__(128) void normrow_bf16(const float* __restrict__ w,
                                                    unsigned short* __restrict__ o, int K) {
  __shared__ float red[128];
  int r = blockIdx.x;
  const float* row = w + (size_t)r * K;
  float ss = 0.f;
  for (int k = threadIdx.x; k < K; k += 128) ss += row[k] * row[k];
  red[threadIdx.x] = ss; __syncthreads();
  for (int s = 64; s > 0; s >>= 1) {
    if (threadIdx.x < s) red[threadIdx.x] += red[threadIdx.x + s];
    __syncthreads();
  }
  float inv = 1.0f / (sqrtf(red[0]) + 1e-8f);
  for (int k = threadIdx.x; k < K; k += 128) o[(size_t)r * K + k] = f2bf(row[k] * inv);
}

__global__ __launch_bounds__(128) void normrow_f32(const float* __restrict__ w,
                                                   float* __restrict__ o, int K) {
  __shared__ float red[128];
  int r = blockIdx.x;
  const float* row = w + (size_t)r * K;
  float ss = 0.f;
  for (int k = threadIdx.x; k < K; k += 128) ss += row[k] * row[k];
  red[threadIdx.x] = ss; __syncthreads();
  for (int s = 64; s > 0; s >>= 1) {
    if (threadIdx.x < s) red[threadIdx.x] += red[threadIdx.x + s];
    __syncthreads();
  }
  float inv = 1.0f / (sqrtf(red[0]) + 1e-8f);
  for (int k = threadIdx.x; k < K; k += 128) o[(size_t)r * K + k] = row[k] * inv;
}

// ---------------------------------------------------------------------------
// Batched GEMM with bf16 WMMA: O[b,m,n] = sum_k W[m,k] * X[b,k,n]   (N = 2048)
// mode 0: plain store; mode 1: *= aux[b*auxStride+m]; mode 2: residual mp_add
// ---------------------------------------------------------------------------
__global__ __launch_bounds__(256) void gemm_bf16_wmma(
    const unsigned short* __restrict__ Wg,   // [M][K] bf16 (normalized)
    const unsigned short* __restrict__ Xg,   // [B][K][NL] bf16
    float* __restrict__ Og,                  // [B][M][NL]
    int M, int K,
    const float* __restrict__ aux, int auxStride,
    const float* __restrict__ res, int mode) {
  __shared__ unsigned short lW[128 * 40];    // [m][k] padded to 40 halves/row
  __shared__ unsigned short lX[128 * 40];    // [n][k] transposed, padded
  const int tid  = threadIdx.x;
  const int b    = blockIdx.z;
  const int m0   = blockIdx.y * 128;
  const int n0   = blockIdx.x * 128;
  const int w    = tid >> 5;
  const int lane = tid & 31;
  const int half = lane >> 4;
  const int ln   = lane & 15;
  const int mw   = w & 3;    // 4 M-waves * 32 rows
  const int nw   = w >> 2;   // 2 N-waves * 64 cols

  v8f acc[2][4] = {};
  const unsigned short* Xb = Xg + (size_t)b * K * NL;

  for (int kk = 0; kk < K; kk += 32) {
    // stage W tile [128][32]: per-lane async global->LDS (ASYNCcnt tracked)
#pragma unroll
    for (int it = 0; it < 2; it++) {
      int t = tid + it * 256;
      int row = t >> 2, ch = t & 3;
      async_copy_b128(Wg + (size_t)(m0 + row) * K + kk + ch * 8,
                      &lW[row * 40 + ch * 8]);
    }
    // stage X tile transposed -> [n][k] (register transpose + b16 scatter)
#pragma unroll
    for (int it = 0; it < 2; it++) {
      int t = tid + it * 256;
      int kr = t >> 4, nc = t & 15;
      PackU p;
      p.q = *(const uint4*)(Xb + (size_t)(kk + kr) * NL + n0 + nc * 8);
#pragma unroll
      for (int j = 0; j < 8; j++) lX[(nc * 8 + j) * 40 + kr] = p.s[j];
    }
    if (kk + 32 < K) {   // hint next tiles (global_prefetch_b8)
      __builtin_prefetch(Wg + (size_t)(m0 + (tid >> 1)) * K + kk + 32, 0, 1);
      __builtin_prefetch(Xb + (size_t)(kk + 32 + (tid >> 4)) * NL + n0 + (tid & 15) * 8, 0, 1);
    }
    wait_async0();       // async LDS writes visible before barrier
    __syncthreads();

    FragU fa[2], fb[4];
#pragma unroll
    for (int mf = 0; mf < 2; mf++) {
      int row = mw * 32 + mf * 16 + ln;
      fa[mf].q[0] = *(const uint4*)(&lW[row * 40 + half * 8]);
      fa[mf].q[1] = *(const uint4*)(&lW[row * 40 + 16 + half * 8]);
    }
#pragma unroll
    for (int nf = 0; nf < 4; nf++) {
      int col = nw * 64 + nf * 16 + ln;
      fb[nf].q[0] = *(const uint4*)(&lX[col * 40 + half * 8]);
      fb[nf].q[1] = *(const uint4*)(&lX[col * 40 + 16 + half * 8]);
    }
#pragma unroll
    for (int mf = 0; mf < 2; mf++)
#pragma unroll
      for (int nf = 0; nf < 4; nf++)
        acc[mf][nf] = __builtin_amdgcn_wmma_f32_16x16x32_bf16(
            false, fa[mf].v, false, fb[nf].v, (short)0, acc[mf][nf], false, false);
    __syncthreads();
  }

#pragma unroll
  for (int mf = 0; mf < 2; mf++) {
#pragma unroll
    for (int nf = 0; nf < 4; nf++) {
      int n = n0 + nw * 64 + nf * 16 + ln;
#pragma unroll
      for (int r = 0; r < 8; r++) {
        int m = m0 + mw * 32 + mf * 16 + half * 8 + r;
        float v = acc[mf][nf][r];
        if (mode == 1) {
          v *= aux[b * auxStride + m];
        } else if (mode == 2) {  // mp_add(xn, v, t=0.3)
          float xv = res[((size_t)b * M + m) * NL + n];
          v = 0.9191450f * xv + 0.3939193f * v;
        }
        Og[((size_t)b * M + m) * NL + n] = v;
      }
    }
  }
}

// ---------------------------------------------------------------------------
// RFF embedding: f, c (fp32 ws), u (to d_out tail)
// ---------------------------------------------------------------------------
__global__ __launch_bounds__(256) void embed_kernel(
    const float* __restrict__ tt, const float* __restrict__ freq,
    const float* __restrict__ phase, const float* __restrict__ We,
    const float* __restrict__ Wl, const float* __restrict__ Wu,
    const float* __restrict__ label, float* __restrict__ cbuf,
    float* __restrict__ u_out) {
  __shared__ float sf[4][64];
  int t = threadIdx.x;
  { int b = t >> 6, j = t & 63;
    sf[b][j] = 1.41421356f * __cosf(6.28318531f * (tt[b] * freq[j] + phase[j])); }
  __syncthreads();
  for (int o = t; o < 1024; o += 256) {
    int b = o >> 8, oc = o & 255;
    const float* we = We + oc * 64;
    float ss = 0.f, le = 0.f;
    for (int k = 0; k < 64; k++) { ss += we[k] * we[k]; le += we[k] * sf[b][k]; }
    le *= 1.0f / (sqrtf(ss) + 1e-8f);
    const float* wl = Wl + oc * 5;
    float ss2 = 0.f, ll = 0.f;
    for (int k = 0; k < 5; k++) { ss2 += wl[k] * wl[k]; ll += wl[k] * label[b * 5 + k]; }
    ll *= 1.0f / (sqrtf(ss2) + 1e-8f);
    float m = (le + ll) * 0.70710678f;                      // mp_add t=0.5
    cbuf[b * 256 + oc] = m * sigmf(m) * 1.6778523f;         // mp_silu
  }
  if (t < 4) {
    float ss = 0.f, s = 0.f;
    for (int k = 0; k < 64; k++) { ss += Wu[k] * Wu[k]; s += Wu[k] * sf[t][k]; }
    u_out[t] = s / (sqrtf(ss) + 1e-8f);
  }
}

// cm[i][b][j] = gain[i] * dot(c[b], wnorm(projc[i][j])) + 1
__global__ __launch_bounds__(256) void cm_kernel(const float* __restrict__ projc,
                                                 const float* __restrict__ gain,
                                                 const float* __restrict__ cbuf,
                                                 float* __restrict__ cm) {
  int t = blockIdx.x * 256 + threadIdx.x;
  if (t >= 8 * 1024) return;
  int i = t >> 10, j = t & 1023;
  const float* row = projc + (size_t)t * 256;
  float ss = 0.f;
  for (int k = 0; k < 256; k++) ss += row[k] * row[k];
  float inv = 1.0f / (sqrtf(ss) + 1e-8f);
  float g = gain[i];
  for (int b = 0; b < 4; b++) {
    const float* cb = cbuf + b * 256;
    float d = 0.f;
    for (int k = 0; k < 256; k++) d += row[k] * cb[k];
    cm[((size_t)i * 4 + b) * 1024 + j] = g * d * inv + 1.0f;
  }
}

// h = conv1x1(mp_cat(x, ones), W_h)
__global__ __launch_bounds__(256) void input_conv(const float* __restrict__ x,
                                                  const float* __restrict__ Whn,
                                                  float* __restrict__ h) {
  size_t idx = (size_t)blockIdx.x * 256 + threadIdx.x;
  if (idx >= (size_t)4 * 256 * NL) return;
  int l = idx % NL;
  int o = (idx / NL) % 256;
  int b = idx / ((size_t)256 * NL);
  const float* wr = Whn + o * 33;
  const float* xb = x + (size_t)b * 32 * NL + l;
  float s = wr[32] * 4.0620192f;                       // ones * (t/sqrt(Nb))*C
  for (int ci = 0; ci < 32; ci++) s += wr[ci] * xb[(size_t)ci * NL] * 0.71806146f;
  h[idx] = s;
}

// pixel_norm -> xn (fp32) ; hh = mp_cat(mp_silu(xn), a) -> bf16
__global__ __launch_bounds__(256) void pixnorm_hh(const float* __restrict__ h,
                                                  const float* __restrict__ a,
                                                  float* __restrict__ xn,
                                                  unsigned short* __restrict__ hhb) {
  __shared__ float red[256];
  int l = blockIdx.x, b = blockIdx.y, c = threadIdx.x;
  float hv = h[((size_t)b * 256 + c) * NL + l];
  red[c] = hv * hv; __syncthreads();
  for (int s = 128; s > 0; s >>= 1) {
    if (c < s) red[c] += red[c + s];
    __syncthreads();
  }
  float rs = rsqrtf(red[0] * (1.0f / 256.0f) + 1e-4f);
  float xv = hv * rs;
  xn[((size_t)b * 256 + c) * NL + l] = xv;
  float sl = xv * sigmf(xv) * 1.6778523f;
  hhb[((size_t)b * 384 + c) * NL + l] = f2bf(sl * 0.8660254f);   // (0.5/sqrt(256))*sqrt(768)
  if (c < 128) {
    float av = a[((size_t)b * 128 + c) * NL + l];
    hhb[((size_t)b * 384 + 256 + c) * NL + l] = f2bf(av * 1.2247449f);
  }
}

// depthwise conv3 (normalized) + mp_silu -> bf16
__global__ __launch_bounds__(256) void dwconv_silu(const float* __restrict__ hg,
                                                   const float* __restrict__ dwn,
                                                   unsigned short* __restrict__ h1b,
                                                   int layer) {
  size_t idx = (size_t)blockIdx.x * 256 + threadIdx.x;
  if (idx >= (size_t)4 * 512 * NL) return;
  int l = idx % NL;
  int c = (idx / NL) % 512;
  int b = idx / ((size_t)512 * NL);
  const float* src = hg + ((size_t)b * 1024 + c) * NL;
  const float* wv  = dwn + ((size_t)layer * 512 + c) * 3;
  float v = wv[1] * src[l];
  if (l > 0)      v += wv[0] * src[l - 1];
  if (l < NL - 1) v += wv[2] * src[l + 1];
  h1b[idx] = f2bf(v * sigmf(v) * 1.6778523f);
}

// minGRU fwd+bwd scans; comb = mp_add(fwd,bwd) * mp_silu(g) -> bf16
__global__ __launch_bounds__(256) void gru_scan(float* __restrict__ gf,
                                                const float* __restrict__ gb,
                                                const float* __restrict__ hg,
                                                unsigned short* __restrict__ comb) {
  int t = blockIdx.x * 256 + threadIdx.x;
  if (t >= 4 * 512) return;
  int b = t / 512, c = t % 512;
  const float* zf = gf + ((size_t)b * 1024 + c) * NL;
  float*       hf = gf + ((size_t)b * 1024 + 512 + c) * NL;   // overwritten with fwd
  float s = 0.f;
  for (int l = 0; l < NL; l++) {
    float z = sigmf(zf[l]);
    s = (1.0f - z) * s + z * hf[l];
    hf[l] = s;
  }
  const float* zb = gb + ((size_t)b * 1024 + c) * NL;
  const float* hb = gb + ((size_t)b * 1024 + 512 + c) * NL;
  const float* gg = hg + ((size_t)b * 1024 + 512 + c) * NL;
  unsigned short* cb = comb + ((size_t)b * 512 + c) * NL;
  float sb = 0.f;
  for (int l = NL - 1; l >= 0; l--) {
    float z = sigmf(zb[l]);
    sb = (1.0f - z) * sb + z * hb[l];
    float mix = (hf[l] + sb) * 0.70710678f;
    float gv = gg[l];
    cb[l] = f2bf(mix * gv * sigmf(gv) * 1.6778523f);
  }
}

// o = out_gain * conv1x1(h, wnorm(Wo))
__global__ __launch_bounds__(256) void final_out(const float* __restrict__ h,
                                                 const float* __restrict__ Won,
                                                 const float* __restrict__ gain,
                                                 float* __restrict__ o) {
  size_t idx = (size_t)blockIdx.x * 256 + threadIdx.x;
  if (idx >= (size_t)4 * 32 * NL) return;
  int l = idx % NL;
  int m = (idx / NL) % 32;
  int b = idx / ((size_t)32 * NL);
  const float* wr = Won + m * 256;
  const float* hb = h + (size_t)b * 256 * NL + l;
  float s = 0.f;
  for (int c = 0; c < 256; c++) s += wr[c] * hb[(size_t)c * NL];
  o[idx] = gain[0] * s;
}

// ---------------------------------------------------------------------------
extern "C" void kernel_launch(void* const* d_in, const int* in_sizes, int n_in,
                              void* d_out, int out_size, void* d_ws, size_t ws_size,
                              hipStream_t stream) {
  (void)in_sizes; (void)n_in; (void)out_size; (void)ws_size;
  const float* a_in   = (const float*)d_in[0];
  const float* label  = (const float*)d_in[1];
  const float* x      = (const float*)d_in[2];
  const float* t_in   = (const float*)d_in[3];
  const float* freq   = (const float*)d_in[4];
  const float* phase  = (const float*)d_in[5];
  const float* W_e    = (const float*)d_in[6];
  const float* W_u    = (const float*)d_in[7];
  const float* W_lab  = (const float*)d_in[8];
  const float* W_h    = (const float*)d_in[9];
  const float* hg_W   = (const float*)d_in[10];
  const float* projc  = (const float*)d_in[11];
  const float* projg  = (const float*)d_in[12];
  const float* dw_W   = (const float*)d_in[13];
  const float* gruf_W = (const float*)d_in[14];
  const float* grub_W = (const float*)d_in[15];
  const float* out_W  = (const float*)d_in[16];
  const float* Wo     = (const float*)d_in[17];
  const float* out_g  = (const float*)d_in[18];
  float* out = (float*)d_out;

  char* ws = (char*)d_ws;
  size_t off = 0;
  auto alloc = [&](size_t bytes) -> void* {
    void* p = ws + off;
    off += (bytes + 255) & ~(size_t)255;
    return p;
  };
  unsigned short* hgb   = (unsigned short*)alloc(8ull * 1024 * 384 * 2);
  unsigned short* grufb = (unsigned short*)alloc(8ull * 1024 * 512 * 2);
  unsigned short* grubb = (unsigned short*)alloc(8ull * 1024 * 512 * 2);
  unsigned short* outwb = (unsigned short*)alloc(8ull * 256 * 512 * 2);
  float* whn   = (float*)alloc(256ull * 33 * 4);
  float* won   = (float*)alloc(32ull * 256 * 4);
  float* dwn   = (float*)alloc(8ull * 512 * 3 * 4);
  float* cbuf  = (float*)alloc(4ull * 256 * 4);
  float* cmw   = (float*)alloc(8ull * 4 * 1024 * 4);
  float* hbuf  = (float*)alloc(4ull * 256 * NL * 4);
  float* xnbuf = (float*)alloc(4ull * 256 * NL * 4);
  unsigned short* hhb = (unsigned short*)alloc(4ull * 384 * NL * 2);
  float* hgout = (float*)alloc(4ull * 1024 * NL * 4);
  unsigned short* h1b = (unsigned short*)alloc(4ull * 512 * NL * 2);  // reused as comb
  float* gfout = (float*)alloc(4ull * 1024 * NL * 4);
  float* gbout = (float*)alloc(4ull * 1024 * NL * 4);

  // weight prep
  normrow_bf16<<<8 * 1024, 128, 0, stream>>>(hg_W,   hgb,   384);
  normrow_bf16<<<8 * 1024, 128, 0, stream>>>(gruf_W, grufb, 512);
  normrow_bf16<<<8 * 1024, 128, 0, stream>>>(grub_W, grubb, 512);
  normrow_bf16<<<8 * 256,  128, 0, stream>>>(out_W,  outwb, 512);
  normrow_f32 <<<256,      128, 0, stream>>>(W_h,  whn, 33);
  normrow_f32 <<<32,       128, 0, stream>>>(Wo,   won, 256);
  normrow_f32 <<<8 * 512,  128, 0, stream>>>(dw_W, dwn, 3);

  // embedding / conditioning
  embed_kernel<<<1, 256, 0, stream>>>(t_in, freq, phase, W_e, W_lab, W_u, label,
                                      cbuf, out + 4ull * 32 * NL);
  cm_kernel<<<32, 256, 0, stream>>>(projc, projg, cbuf, cmw);

  // input conv
  input_conv<<<(4 * 256 * NL) / 256, 256, 0, stream>>>(x, whn, hbuf);

  for (int i = 0; i < 8; i++) {
    pixnorm_hh<<<dim3(NL, 4), 256, 0, stream>>>(hbuf, a_in, xnbuf, hhb);
    // hg conv with cm conditioning: 1024 x 384
    gemm_bf16_wmma<<<dim3(16, 8, 4), 256, 0, stream>>>(
        hgb + (size_t)i * 1024 * 384, hhb, hgout, 1024, 384,
        cmw + (size_t)i * 4 * 1024, 1024, nullptr, 1);
    dwconv_silu<<<(4 * 512 * NL) / 256, 256, 0, stream>>>(hgout, dwn, h1b, i);
    // gru convs: 1024 x 512
    gemm_bf16_wmma<<<dim3(16, 8, 4), 256, 0, stream>>>(
        grufb + (size_t)i * 1024 * 512, h1b, gfout, 1024, 512, nullptr, 0, nullptr, 0);
    gemm_bf16_wmma<<<dim3(16, 8, 4), 256, 0, stream>>>(
        grubb + (size_t)i * 1024 * 512, h1b, gbout, 1024, 512, nullptr, 0, nullptr, 0);
    // scans + gate multiply (comb reuses h1b storage; GEMMs above already consumed it)
    gru_scan<<<8, 256, 0, stream>>>(gfout, gbout, hgout, h1b);
    // out conv with residual mp_add(xn, ., t=0.3): 256 x 512
    gemm_bf16_wmma<<<dim3(16, 2, 4), 256, 0, stream>>>(
        outwb + (size_t)i * 256 * 512, h1b, hbuf, 256, 512, nullptr, 0, xnbuf, 2);
  }

  final_out<<<(4 * 32 * NL) / 256, 256, 0, stream>>>(hbuf, won, out_g, out);
}
